// StructureEncoder_18056042512610
// MI455X (gfx1250) — matrix operand order repaired and checked by reading, not compiled, not run on hardware
//
#include <hip/hip_runtime.h>
#include <hip/hip_bf16.h>

typedef __attribute__((ext_vector_type(16))) _Float16 v16h;
typedef __attribute__((ext_vector_type(8)))  float    v8f;

constexpr int Zb   = 4;
constexpr int Nn   = 2048;
constexpr int Dd   = 128;
constexpr int TK   = 32;
constexpr int RBFD = 256;                 // 4*4*16
constexpr int CIN  = 3 * Dd;              // 384
constexpr long long NEDGE = (long long)Zb * Nn * TK;   // 262144
constexpr float EPSE    = 1e-12f;
constexpr float SPREAD2 = 1.25f * 1.25f;  // ((22-2)/16)^2
constexpr float LNEPS   = 1e-5f;

// ---------------------------------------------------------------- helpers
__device__ __forceinline__ float gelu_f(float x) {
  return 0.5f * x * (1.0f + erff(x * 0.70710678118654752f));
}
__device__ __forceinline__ v8f vzero8() {
  v8f z = {0.f, 0.f, 0.f, 0.f, 0.f, 0.f, 0.f, 0.f};
  return z;
}
__device__ __forceinline__ v8f wmma16(v16h a, v16h b, v8f c) {
  return __builtin_amdgcn_wmma_f32_16x16x32_f16(false, a, false, b, (short)0, c,
                                                false, false);
}

// A fragment: 16x32 f16 tile from row-major LDS, leading dim ld.
// Lane L: row = L&15, half = L>>4; elems 0..7 -> K = half*8+0..7,
// elems 8..15 -> K = 16 + half*8 + 0..7  (per CDNA5 ISA 16-bit A layout)
__device__ __forceinline__ v16h load_a_lds(const _Float16* t, int row_base,
                                           int k_base, int ld) {
  int lane = threadIdx.x & 31;
  const _Float16* p =
      t + (size_t)(row_base + (lane & 15)) * ld + k_base + (lane >> 4) * 8;
  v16h a;
#pragma unroll
  for (int r = 0; r < 8; ++r) a[r] = p[r];
#pragma unroll
  for (int r = 0; r < 8; ++r) a[8 + r] = p[16 + r];
  return a;
}

// B fragment from pre-swizzled f16 weights: tile (kc,nt) of a (K x N) matrix
// stored as 512 contiguous halves, lane-major (lane*16 + kk).  Two aligned
// global_load_b128 per fragment, zero conversion ALU in the hot loop.
__device__ __forceinline__ v16h load_b_swz(const _Float16* wz, int kc, int nt,
                                           int tn /* = N/16 */) {
  const _Float16* p =
      wz + (((size_t)kc * tn + nt) << 9) + ((threadIdx.x & 31) << 4);
  v16h b;
#pragma unroll
  for (int kk = 0; kk < 16; ++kk) b[kk] = p[kk];
  return b;
}

// ---------------------------------------------------------------- weight prep
// f32 (K x N) row-major  ->  f16 fragment-swizzled (tile-major, lane-major)
__global__ void wprep_kernel(const float* __restrict__ src,
                             _Float16* __restrict__ dst, int K, int N) {
  int idx = blockIdx.x * blockDim.x + threadIdx.x;
  if (idx >= K * N) return;
  int k = idx / N, n = idx - k * N;
  int kc = k >> 5, kr = k & 31;
  int nt = n >> 4, nc = n & 15;
  int lane = (((kr >> 3) & 1) << 4) + nc;  // half*16 + col
  int kk = (kr & 7) + ((kr >> 4) << 3);
  dst[(((size_t)kc * (N >> 4) + nt) << 9) + (lane << 4) + kk] =
      (_Float16)src[idx];
}

// ---------------------------------------------------------------- init
__global__ void init_kernel(float* V, _Float16* Vh) {
  long long idx = (long long)blockIdx.x * blockDim.x + threadIdx.x;
  if (idx < (long long)Zb * Nn * Dd) {
    V[idx] = 0.f;
    Vh[idx] = (_Float16)0.f;
  }
}

// ---------------------------------------------------------------- knn
__global__ void knn_kernel(const float* __restrict__ C,
                           const unsigned char* __restrict__ vm,
                           int* __restrict__ nbrs,
                           unsigned char* __restrict__ emask) {
  int t = blockIdx.x * blockDim.x + threadIdx.x;
  if (t >= Zb * Nn) return;
  int z = t / Nn, i = t - z * Nn;
  size_t cbase = (((size_t)z * Nn + i) * 4 + 1) * 3;
  float xi = C[cbase + 0], yi = C[cbase + 1], zi = C[cbase + 2];
  float bd[TK];
  int bi[TK];
#pragma unroll
  for (int k = 0; k < TK; ++k) { bd[k] = 3.0e38f; bi[k] = -1; }
  bool vi = vm[(size_t)z * Nn + i] != 0;
  for (int j = 0; j < Nn; ++j) {
    size_t jb = (((size_t)z * Nn + j) * 4 + 1) * 3;
    float dx = xi - C[jb], dy = yi - C[jb + 1], dz = zi - C[jb + 2];
    float sq = dx * dx + dy * dy + dz * dz;
    if (sq == 0.f) continue;
    if (!vm[(size_t)z * Nn + j]) continue;
    float d = sqrtf(sq + EPSE);
    if (d < bd[TK - 1]) {
      int p = TK - 1;
      while (p > 0 && bd[p - 1] > d) {
        bd[p] = bd[p - 1]; bi[p] = bi[p - 1]; --p;
      }
      bd[p] = d; bi[p] = j;
    }
  }
  for (int k = 0; k < TK; ++k) {
    bool ok = vi && (bi[k] >= 0);
    nbrs[(size_t)t * TK + k] = ok ? bi[k] : i;
    emask[(size_t)t * TK + k] = ok ? 1 : 0;
  }
}

// ---------------------------------------------------------------- edges
// 64 edges / block, 256 threads (8 waves).  RBF -> LN -> WMMA 256x128 proj.
__global__ void __launch_bounds__(256)
edge_kernel(const float* __restrict__ C, const int* __restrict__ nbrs,
            const float* __restrict__ lng, const float* __restrict__ lnb,
            const _Float16* __restrict__ Wz, const float* __restrict__ bias,
            float* __restrict__ E, _Float16* __restrict__ Eh) {
  extern __shared__ char smem[];
  float* dp = (float*)smem;                              // 64*16 f32
  _Float16* A = (_Float16*)(smem + 64 * 16 * 4);         // 64*256 f16
  float* redS = (float*)(smem + 64 * 16 * 4 + 64 * RBFD * 2);  // 64*4
  float* redQ = redS + 64 * 4;                                  // 64*4
  int tid = threadIdx.x;
  size_t eb = (size_t)blockIdx.x * 64;

  for (int idx = tid; idx < 64 * 16; idx += 256) {
    int el = idx >> 4, p = idx & 15;
    int a = p >> 2, b = p & 3;
    size_t e = eb + el;
    int z = (int)(e / ((size_t)Nn * TK));
    int i = (int)((e / TK) % Nn);
    int j = nbrs[e];
    const float* pi = C + (((size_t)z * Nn + i) * 4 + a) * 3;
    const float* pj = C + (((size_t)z * Nn + j) * 4 + b) * 3;
    float dx = pi[0] - pj[0], dy = pi[1] - pj[1], dz = pi[2] - pj[2];
    dp[idx] = sqrtf(dx * dx + dy * dy + dz * dz + EPSE);
  }
  __syncthreads();
  for (int idx = tid; idx < 64 * RBFD; idx += 256) {
    int el = idx >> 8, f = idx & 255;
    int p = f >> 4, r = f & 15;
    float d = dp[el * 16 + p];
    float c = 2.0f + (20.0f / 15.0f) * (float)r;
    float u = d - c;
    A[idx] = (_Float16)expf(-u * u / SPREAD2);
  }
  __syncthreads();
  {  // LN stats: 4 threads per edge row of 256
    int el = tid >> 2, part = tid & 3;
    float s = 0.f, q = 0.f;
    for (int f = part * 64; f < part * 64 + 64; ++f) {
      float v = (float)A[el * RBFD + f];
      s += v; q += v * v;
    }
    redS[el * 4 + part] = s; redQ[el * 4 + part] = q;
  }
  __syncthreads();
  if (tid < 64) {
    float s = redS[tid * 4] + redS[tid * 4 + 1] + redS[tid * 4 + 2] + redS[tid * 4 + 3];
    float q = redQ[tid * 4] + redQ[tid * 4 + 1] + redQ[tid * 4 + 2] + redQ[tid * 4 + 3];
    float m = s / RBFD;
    redS[tid * 4] = m;
    redQ[tid * 4] = rsqrtf(q / RBFD - m * m + LNEPS);
  }
  __syncthreads();
  for (int idx = tid; idx < 64 * RBFD; idx += 256) {
    int el = idx >> 8, f = idx & 255;
    float v = (float)A[idx];
    v = (v - redS[el * 4]) * redQ[el * 4] * lng[f] + lnb[f];
    A[idx] = (_Float16)v;
  }
  __syncthreads();

  int w = tid >> 5;  // wave owns columns [16w,16w+16)
  v8f acc[4];
#pragma unroll
  for (int mt = 0; mt < 4; ++mt) acc[mt] = vzero8();
  for (int kc = 0; kc < RBFD / 32; ++kc) {
    v16h bf = load_b_swz(Wz, kc, w, Dd / 16);
#pragma unroll
    for (int mt = 0; mt < 4; ++mt) {
      v16h af = load_a_lds(A, mt * 16, kc * 32, RBFD);
      acc[mt] = wmma16(af, bf, acc[mt]);
    }
  }
  int lane = tid & 31;
  int ncol = w * 16 + (lane & 15);
  int mo = (lane >> 4) * 8;
  float bb = bias[ncol];
#pragma unroll
  for (int mt = 0; mt < 4; ++mt) {
#pragma unroll
    for (int r = 0; r < 8; ++r) {
      float v = acc[mt][r] + bb;
      size_t off = (eb + mt * 16 + mo + r) * Dd + ncol;
      E[off] = v;
      Eh[off] = (_Float16)v;
    }
  }
}

// ---------------------------------------------------------------- message MLP
// 64 edges (= exactly 2 nodes) / block.  A=[Vi|Vj|E] -> gelu GEMM x2 -> GEMM.
// mode 0: per-node sum over K into Msum.   mode 1: E = LN(E + out).
// The E-slice of the A tile is contiguous (64 rows x 256B) -> moved with
// GLOBAL_LOAD_ASYNC_TO_LDS_B128 (ASYNCcnt), overlapping the indexed Vi/Vj fill.
__global__ void __launch_bounds__(256)
msg_kernel(const _Float16* __restrict__ Vh,
           float* __restrict__ E, _Float16* __restrict__ Eh,
           const int* __restrict__ nbrs, const unsigned char* __restrict__ emask,
           const _Float16* __restrict__ W1z, const float* __restrict__ B1,
           const _Float16* __restrict__ W2z, const float* __restrict__ B2,
           const _Float16* __restrict__ W3z, const float* __restrict__ B3,
           const float* __restrict__ lng, const float* __restrict__ lnb,
           float* __restrict__ Msum, int edge_mode) {
  extern __shared__ char smem[];
  _Float16* A  = (_Float16*)smem;                       // 64*384 f16
  _Float16* H1 = (_Float16*)(smem + 64 * CIN * 2);      // 64*128 f16
  _Float16* H2 = H1 + 64 * Dd;                          // 64*128 f16
  float* OutT  = (float*)(smem + 64 * CIN * 2 + 2 * 64 * Dd * 2);  // 64*128 f32
  float* redS  = OutT + 64 * Dd;                        // 64*4
  float* redQ  = redS + 64 * 4;                         // 64*4
  int* jrow    = (int*)(redQ + 64 * 4);                 // 64
  int tid = threadIdx.x;
  size_t eb = (size_t)blockIdx.x * 64;
  int z = (int)(eb / ((size_t)Nn * TK));

  if (tid < 64) jrow[tid] = nbrs[eb + tid];

  // async copy of the contiguous E slice into A[:, 256:384]
  for (int idx = tid; idx < 64 * 16; idx += 256) {
    int row = idx >> 4, chunk = idx & 15;
    unsigned ldsa = (unsigned)(size_t)(&A[(size_t)row * CIN + 2 * Dd]) +
                    (unsigned)(chunk * 16);
    unsigned long long ga =
        (unsigned long long)(const void*)(Eh + (eb + row) * Dd) +
        (unsigned long long)(chunk * 16);
    asm volatile("global_load_async_to_lds_b128 %0, %1, off"
                 :
                 : "v"(ldsa), "v"(ga)
                 : "memory");
  }
  // indexed Vi / Vj gather for A[:, 0:256]
  for (int idx = tid; idx < 64 * 2 * Dd; idx += 256) {
    int el = idx >> 8, f = idx & 255;
    size_t e = eb + el;
    int i = (int)((e / TK) % Nn);
    _Float16 v;
    if (f < Dd) v = Vh[((size_t)z * Nn + i) * Dd + f];
    else        v = Vh[((size_t)z * Nn + jrow[el]) * Dd + (f - Dd)];
    A[(size_t)el * CIN + f] = v;
  }
  asm volatile("s_wait_asynccnt 0x0" ::: "memory");
  __syncthreads();

  int w = tid >> 5;
  int lane = tid & 31;
  int ncol = w * 16 + (lane & 15);
  int mo = (lane >> 4) * 8;

  {  // GEMM1: 64x384 @ 384x128 -> gelu -> H1
    v8f acc[4];
#pragma unroll
    for (int mt = 0; mt < 4; ++mt) acc[mt] = vzero8();
    for (int kc = 0; kc < CIN / 32; ++kc) {
      v16h bf = load_b_swz(W1z, kc, w, Dd / 16);
#pragma unroll
      for (int mt = 0; mt < 4; ++mt)
        acc[mt] = wmma16(load_a_lds(A, mt * 16, kc * 32, CIN), bf, acc[mt]);
    }
    float bb = B1[ncol];
#pragma unroll
    for (int mt = 0; mt < 4; ++mt)
#pragma unroll
      for (int r = 0; r < 8; ++r)
        H1[(mt * 16 + mo + r) * Dd + ncol] = (_Float16)gelu_f(acc[mt][r] + bb);
  }
  __syncthreads();
  {  // GEMM2: 64x128 @ 128x128 -> gelu -> H2
    v8f acc[4];
#pragma unroll
    for (int mt = 0; mt < 4; ++mt) acc[mt] = vzero8();
    for (int kc = 0; kc < Dd / 32; ++kc) {
      v16h bf = load_b_swz(W2z, kc, w, Dd / 16);
#pragma unroll
      for (int mt = 0; mt < 4; ++mt)
        acc[mt] = wmma16(load_a_lds(H1, mt * 16, kc * 32, Dd), bf, acc[mt]);
    }
    float bb = B2[ncol];
#pragma unroll
    for (int mt = 0; mt < 4; ++mt)
#pragma unroll
      for (int r = 0; r < 8; ++r)
        H2[(mt * 16 + mo + r) * Dd + ncol] = (_Float16)gelu_f(acc[mt][r] + bb);
  }
  __syncthreads();
  {  // GEMM3: 64x128 @ 128x128 -> + bias, * mask -> OutT (f32)
    v8f acc[4];
#pragma unroll
    for (int mt = 0; mt < 4; ++mt) acc[mt] = vzero8();
    for (int kc = 0; kc < Dd / 32; ++kc) {
      v16h bf = load_b_swz(W3z, kc, w, Dd / 16);
#pragma unroll
      for (int mt = 0; mt < 4; ++mt)
        acc[mt] = wmma16(load_a_lds(H2, mt * 16, kc * 32, Dd), bf, acc[mt]);
    }
    float bb = B3[ncol];
#pragma unroll
    for (int mt = 0; mt < 4; ++mt)
#pragma unroll
      for (int r = 0; r < 8; ++r) {
        int row = mt * 16 + mo + r;
        float m = emask[eb + row] ? 1.f : 0.f;
        OutT[row * Dd + ncol] = (acc[mt][r] + bb) * m;
      }
  }
  __syncthreads();

  if (!edge_mode) {
    // sum over K=32 edges for each of the 2 nodes in this tile
    int col = tid & 127, half = tid >> 7;
    float s = 0.f;
#pragma unroll
    for (int r = 0; r < TK; ++r) s += OutT[(half * TK + r) * Dd + col];
    size_t e = eb + half * TK;
    int i = (int)((e / TK) % Nn);
    Msum[((size_t)z * Nn + i) * Dd + col] = s;
  } else {
    // E = LN(E + out)
    int el = tid >> 2, part = tid & 3;
    float s = 0.f, q = 0.f;
    for (int f = part * 32; f < part * 32 + 32; ++f) {
      float v = E[(eb + el) * Dd + f] + OutT[el * Dd + f];
      s += v; q += v * v;
    }
    redS[el * 4 + part] = s; redQ[el * 4 + part] = q;
    __syncthreads();
    if (tid < 64) {
      float ss = redS[tid * 4] + redS[tid * 4 + 1] + redS[tid * 4 + 2] + redS[tid * 4 + 3];
      float qq = redQ[tid * 4] + redQ[tid * 4 + 1] + redQ[tid * 4 + 2] + redQ[tid * 4 + 3];
      float m = ss / Dd;
      redS[tid * 4] = m;
      redQ[tid * 4] = rsqrtf(qq / Dd - m * m + LNEPS);
    }
    __syncthreads();
    for (int f = part * 32; f < part * 32 + 32; ++f) {
      size_t off = (eb + el) * Dd + f;
      float v = E[off] + OutT[el * Dd + f];
      v = (v - redS[el * 4]) * redQ[el * 4] * lng[f] + lnb[f];
      E[off] = v;
      Eh[off] = (_Float16)v;
    }
  }
}

// ---------------------------------------------------------------- node LN
// one wave (32 lanes) per node; wave32 shfl reduction
__global__ void __launch_bounds__(256)
node_ln_kernel(float* __restrict__ V, _Float16* __restrict__ Vh,
               const float* __restrict__ Msum, const float* __restrict__ g,
               const float* __restrict__ b) {
  int node = blockIdx.x * 8 + (threadIdx.x >> 5);
  int lane = threadIdx.x & 31;
  size_t base = (size_t)node * Dd;
  float x[4], s = 0.f, q = 0.f;
#pragma unroll
  for (int jj = 0; jj < 4; ++jj) {
    int c = lane * 4 + jj;
    x[jj] = V[base + c] + Msum[base + c];
    s += x[jj]; q += x[jj] * x[jj];
  }
#pragma unroll
  for (int o = 16; o > 0; o >>= 1) {
    s += __shfl_xor(s, o, 32);
    q += __shfl_xor(q, o, 32);
  }
  float m = s / Dd;
  float rs = rsqrtf(q / Dd - m * m + LNEPS);
#pragma unroll
  for (int jj = 0; jj < 4; ++jj) {
    int c = lane * 4 + jj;
    float v = (x[jj] - m) * rs * g[c] + b[c];
    V[base + c] = v;
    Vh[base + c] = (_Float16)v;
  }
}

// ---------------------------------------------------------------- FFN
// 32 nodes / block: V@W1(128x512) gelu @W2(512x128), residual + LN
__global__ void __launch_bounds__(256)
ffn_kernel(float* __restrict__ V, _Float16* __restrict__ Vh,
           const _Float16* __restrict__ W1z, const float* __restrict__ B1,
           const _Float16* __restrict__ W2z, const float* __restrict__ B2,
           const float* __restrict__ g, const float* __restrict__ b) {
  extern __shared__ char smem[];
  _Float16* A = (_Float16*)smem;                     // 32*128 f16
  _Float16* H = A + 32 * Dd;                         // 32*512 f16
  float* OutT = (float*)(smem + 32 * Dd * 2 + 32 * 512 * 2);  // 32*128 f32
  float* redS = OutT + 32 * Dd;                      // 32*8
  float* redQ = redS + 32 * 8;                       // 32*8
  int tid = threadIdx.x;
  size_t nb = (size_t)blockIdx.x * 32;

  for (int idx = tid; idx < 32 * Dd; idx += 256) {
    int row = idx >> 7, c = idx & 127;
    A[idx] = Vh[(nb + row) * Dd + c];
  }
  __syncthreads();

  int w = tid >> 5;
  int lane = tid & 31;
  int mo = (lane >> 4) * 8;

  {  // GEMM1 -> gelu -> H ; wave w owns columns [64w, 64w+64)
    v8f acc[2][4];
#pragma unroll
    for (int mt = 0; mt < 2; ++mt)
#pragma unroll
      for (int t = 0; t < 4; ++t) acc[mt][t] = vzero8();
    for (int kc = 0; kc < Dd / 32; ++kc) {
      v16h a0 = load_a_lds(A, 0, kc * 32, Dd);
      v16h a1 = load_a_lds(A, 16, kc * 32, Dd);
#pragma unroll
      for (int t = 0; t < 4; ++t) {
        v16h bf = load_b_swz(W1z, kc, w * 4 + t, 512 / 16);
        acc[0][t] = wmma16(a0, bf, acc[0][t]);
        acc[1][t] = wmma16(a1, bf, acc[1][t]);
      }
    }
#pragma unroll
    for (int t = 0; t < 4; ++t) {
      int col = w * 64 + t * 16 + (lane & 15);
      float bb = B1[col];
#pragma unroll
      for (int mt = 0; mt < 2; ++mt)
#pragma unroll
        for (int r = 0; r < 8; ++r)
          H[(mt * 16 + mo + r) * 512 + col] =
              (_Float16)gelu_f(acc[mt][t][r] + bb);
    }
  }
  __syncthreads();
  {  // GEMM2: 32x512 @ 512x128 -> OutT
    v8f a2[2] = {vzero8(), vzero8()};
    for (int kc = 0; kc < 512 / 32; ++kc) {
      v16h bf = load_b_swz(W2z, kc, w, Dd / 16);
      a2[0] = wmma16(load_a_lds(H, 0, kc * 32, 512), bf, a2[0]);
      a2[1] = wmma16(load_a_lds(H, 16, kc * 32, 512), bf, a2[1]);
    }
    int ncol = w * 16 + (lane & 15);
    float bb = B2[ncol];
#pragma unroll
    for (int mt = 0; mt < 2; ++mt)
#pragma unroll
      for (int r = 0; r < 8; ++r)
        OutT[(mt * 16 + mo + r) * Dd + ncol] = a2[mt][r] + bb;
  }
  __syncthreads();

  // residual + LN: 8 partial-threads per 128-wide row
  int row = tid >> 3, part = tid & 7;
  float s = 0.f, q = 0.f;
  for (int f = part * 16; f < part * 16 + 16; ++f) {
    float v = V[(nb + row) * Dd + f] + OutT[row * Dd + f];
    s += v; q += v * v;
  }
  redS[row * 8 + part] = s; redQ[row * 8 + part] = q;
  __syncthreads();
  if (tid < 32) {
    float ss = 0.f, qq = 0.f;
#pragma unroll
    for (int p = 0; p < 8; ++p) { ss += redS[tid * 8 + p]; qq += redQ[tid * 8 + p]; }
    float m = ss / Dd;
    redS[tid * 8] = m;
    redQ[tid * 8] = rsqrtf(qq / Dd - m * m + LNEPS);
  }
  __syncthreads();
  for (int f = part * 16; f < part * 16 + 16; ++f) {
    size_t off = (nb + row) * Dd + f;
    float v = V[off] + OutT[row * Dd + f];
    v = (v - redS[row * 8]) * redQ[row * 8] * g[f] + b[f];
    V[off] = v;
    Vh[off] = (_Float16)v;
  }
}

// ---------------------------------------------------------------- pack
__global__ void pack_kernel(const float* __restrict__ V,
                            const float* __restrict__ E,
                            const int* __restrict__ nbrs,
                            float* __restrict__ out) {
  const long long nv = (long long)Zb * Nn * Dd;
  const long long ne = NEDGE * Dd;
  const long long total = nv + ne + NEDGE;
  for (long long idx = (long long)blockIdx.x * blockDim.x + threadIdx.x;
       idx < total; idx += (long long)gridDim.x * blockDim.x) {
    if (idx < nv)            out[idx] = V[idx];
    else if (idx < nv + ne)  out[idx] = E[idx - nv];
    else                     out[idx] = __int_as_float(nbrs[idx - nv - ne]);
  }
}

// ---------------------------------------------------------------- launch
extern "C" void kernel_launch(void* const* d_in, const int* in_sizes, int n_in,
                              void* d_out, int out_size, void* d_ws,
                              size_t ws_size, hipStream_t stream) {
  (void)in_sizes; (void)n_in; (void)out_size; (void)ws_size;
  const float* C = (const float*)d_in[0];
  const unsigned char* vm = (const unsigned char*)d_in[1];
  const float* elng = (const float*)d_in[2];
  const float* elnb = (const float*)d_in[3];
  const float* eW = (const float*)d_in[4];
  const float* eB = (const float*)d_in[5];
  const float* nmW1 = (const float*)d_in[6];  const float* nmB1 = (const float*)d_in[7];
  const float* nmW2 = (const float*)d_in[8];  const float* nmB2 = (const float*)d_in[9];
  const float* nmW3 = (const float*)d_in[10]; const float* nmB3 = (const float*)d_in[11];
  const float* nmnG = (const float*)d_in[12]; const float* nmnB = (const float*)d_in[13];
  const float* fW1 = (const float*)d_in[14];  const float* fB1 = (const float*)d_in[15];
  const float* fW2 = (const float*)d_in[16];  const float* fB2 = (const float*)d_in[17];
  const float* fG  = (const float*)d_in[18];  const float* fBb = (const float*)d_in[19];
  const float* emW1 = (const float*)d_in[20]; const float* emB1 = (const float*)d_in[21];
  const float* emW2 = (const float*)d_in[22]; const float* emB2 = (const float*)d_in[23];
  const float* emW3 = (const float*)d_in[24]; const float* emB3 = (const float*)d_in[25];
  const float* emnG = (const float*)d_in[26]; const float* emnB = (const float*)d_in[27];

  char* ws = (char*)d_ws;
  size_t off = 0;
  auto alloc = [&](size_t bytes) -> void* {
    size_t cur = off;
    off += (bytes + 255) & ~(size_t)255;
    return (void*)(ws + cur);
  };
  int* nbrs = (int*)alloc(sizeof(int) * NEDGE);
  unsigned char* emask = (unsigned char*)alloc(NEDGE);
  float* E = (float*)alloc(sizeof(float) * (size_t)NEDGE * Dd);
  _Float16* Eh = (_Float16*)alloc(sizeof(_Float16) * (size_t)NEDGE * Dd);
  float* V = (float*)alloc(sizeof(float) * (size_t)Zb * Nn * Dd);
  _Float16* Vh = (_Float16*)alloc(sizeof(_Float16) * (size_t)Zb * Nn * Dd);
  float* Msum = (float*)alloc(sizeof(float) * (size_t)Zb * Nn * Dd);
  // swizzled f16 weights
  _Float16* eWz  = (_Float16*)alloc(sizeof(_Float16) * RBFD * Dd);
  _Float16* nm1z = (_Float16*)alloc(sizeof(_Float16) * 3 * CIN * Dd);
  _Float16* nm2z = (_Float16*)alloc(sizeof(_Float16) * 3 * Dd * Dd);
  _Float16* nm3z = (_Float16*)alloc(sizeof(_Float16) * 3 * Dd * Dd);
  _Float16* f1z  = (_Float16*)alloc(sizeof(_Float16) * 3 * Dd * 4 * Dd);
  _Float16* f2z  = (_Float16*)alloc(sizeof(_Float16) * 3 * 4 * Dd * Dd);
  _Float16* em1z = (_Float16*)alloc(sizeof(_Float16) * 3 * CIN * Dd);
  _Float16* em2z = (_Float16*)alloc(sizeof(_Float16) * 3 * Dd * Dd);
  _Float16* em3z = (_Float16*)alloc(sizeof(_Float16) * 3 * Dd * Dd);

  // ---- weight prep (tiny; weights stay L2-resident afterwards)
  auto prep = [&](const float* s, _Float16* d, int K, int N) {
    wprep_kernel<<<(K * N + 255) / 256, 256, 0, stream>>>(s, d, K, N);
  };
  prep(eW, eWz, RBFD, Dd);
  for (int l = 0; l < 3; ++l) {
    prep(nmW1 + (size_t)l * CIN * Dd, nm1z + (size_t)l * CIN * Dd, CIN, Dd);
    prep(nmW2 + (size_t)l * Dd * Dd, nm2z + (size_t)l * Dd * Dd, Dd, Dd);
    prep(nmW3 + (size_t)l * Dd * Dd, nm3z + (size_t)l * Dd * Dd, Dd, Dd);
    prep(fW1 + (size_t)l * Dd * 4 * Dd, f1z + (size_t)l * Dd * 4 * Dd, Dd, 4 * Dd);
    prep(fW2 + (size_t)l * 4 * Dd * Dd, f2z + (size_t)l * 4 * Dd * Dd, 4 * Dd, Dd);
    prep(emW1 + (size_t)l * CIN * Dd, em1z + (size_t)l * CIN * Dd, CIN, Dd);
    prep(emW2 + (size_t)l * Dd * Dd, em2z + (size_t)l * Dd * Dd, Dd, Dd);
    prep(emW3 + (size_t)l * Dd * Dd, em3z + (size_t)l * Dd * Dd, Dd, Dd);
  }

  const int EDGE_LDS = 64 * 16 * 4 + 64 * RBFD * 2 + 64 * 4 * 4 * 2;
  const int MSG_LDS = 64 * CIN * 2 + 2 * 64 * Dd * 2 + 64 * Dd * 4 +
                      64 * 4 * 4 * 2 + 64 * 4;
  const int FFN_LDS = 32 * Dd * 2 + 32 * 512 * 2 + 32 * Dd * 4 + 32 * 8 * 4 * 2;

  init_kernel<<<(Zb * Nn * Dd + 255) / 256, 256, 0, stream>>>(V, Vh);
  knn_kernel<<<(Zb * Nn + 255) / 256, 256, 0, stream>>>(C, vm, nbrs, emask);
  edge_kernel<<<(int)(NEDGE / 64), 256, EDGE_LDS, stream>>>(C, nbrs, elng, elnb,
                                                            eWz, eB, E, Eh);
  for (int l = 0; l < 3; ++l) {
    size_t o1 = (size_t)l * CIN * Dd;
    size_t ob = (size_t)l * Dd;
    size_t o2 = (size_t)l * Dd * Dd;
    msg_kernel<<<(int)(NEDGE / 64), 256, MSG_LDS, stream>>>(
        Vh, E, Eh, nbrs, emask, nm1z + o1, nmB1 + ob, nm2z + o2, nmB2 + ob,
        nm3z + o2, nmB3 + ob, nullptr, nullptr, Msum, 0);
    node_ln_kernel<<<Zb * Nn / 8, 256, 0, stream>>>(V, Vh, Msum, nmnG + ob,
                                                    nmnB + ob);
    ffn_kernel<<<Zb * Nn / 32, 256, FFN_LDS, stream>>>(
        V, Vh, f1z + (size_t)l * Dd * 4 * Dd, fB1 + (size_t)l * 4 * Dd,
        f2z + (size_t)l * 4 * Dd * Dd, fB2 + ob, fG + ob, fBb + ob);
    msg_kernel<<<(int)(NEDGE / 64), 256, MSG_LDS, stream>>>(
        Vh, E, Eh, nbrs, emask, em1z + o1, emB1 + ob, em2z + o2, emB2 + ob,
        em3z + o2, emB3 + ob, emnG + ob, emnB + ob, Msum, 1);
  }
  pack_kernel<<<2048, 256, 0, stream>>>(V, E, nbrs, (float*)d_out);
}
